// PolymerGNN_6640019440133
// MI455X (gfx1250) — matrix operand order repaired and checked by reading, not compiled, not run on hardware
//
#include <hip/hip_runtime.h>
#include <hip/hip_bf16.h>
#include <math.h>

typedef __bf16 bf16_t;
typedef __attribute__((ext_vector_type(16))) __bf16 v16bf;
typedef __attribute__((ext_vector_type(8)))  float  v8f;

#define HID 128
#define NHEAD 4
#define CH 32
#define NLAYER 4
#define GDIM 10
#define TOUT 5
#define EPSV 1e-5f
#define NEG_SLOPE 0.2f

// ---- float <-> order-preserving uint (for atomic max on floats) ----
__device__ __forceinline__ unsigned f2ord(float f) {
  unsigned u = __float_as_uint(f);
  return (u & 0x80000000u) ? ~u : (u | 0x80000000u);
}
__device__ __forceinline__ float ord2f(unsigned t) {
  unsigned u = (t & 0x80000000u) ? (t ^ 0x80000000u) : ~t;
  return __uint_as_float(u);
}

// ---------------------------------------------------------------------------
// Generic small dense layer: Out[M][Nout] = In[M][K] @ W[K][Nout] + b, opt relu
// one block per row; input row staged in LDS.
// ---------------------------------------------------------------------------
__global__ void k_lin_small(const float* __restrict__ In, const float* __restrict__ W,
                            const float* __restrict__ bvec, float* __restrict__ Out,
                            int K, int Nout, int relu) {
  extern __shared__ float xs[];
  int n = blockIdx.x;
  for (int i = threadIdx.x; i < K; i += blockDim.x) xs[i] = In[(size_t)n * K + i];
  __syncthreads();
  int t = threadIdx.x;
  if (t >= Nout) return;
  float acc = bvec[t];
  for (int k = 0; k < K; ++k) acc = fmaf(xs[k], W[(size_t)k * Nout + t], acc);
  if (relu) acc = fmaxf(acc, 0.f);
  Out[(size_t)n * Nout + t] = acc;
}

// ---------------------------------------------------------------------------
// Pack f32 weight [K][Nn] (row-major) into per-lane WMMA-B bf16 layout:
// lane l of tile (kt,nt) holds K = kt*32 + (l>=16?16:0) + j (j=0..15),
// column nt*16 + (l&15). 16 contiguous bf16 per lane -> 2x b128 loads in GEMM.
// ---------------------------------------------------------------------------
__global__ void k_pack_w(const float* __restrict__ W, bf16_t* __restrict__ P,
                         int K, int Nn) {
  int kt = blockIdx.x, nt = blockIdx.y, l = threadIdx.x;
  int ntiles = Nn >> 4;
  bf16_t* dst = P + ((size_t)(kt * ntiles + nt) * 32 + l) * 16;
  int kbase = kt * 32 + ((l >> 4) << 4);
  int ncol = nt * 16 + (l & 15);
#pragma unroll
  for (int j = 0; j < 16; ++j) dst[j] = (bf16_t)W[(size_t)(kbase + j) * Nn + ncol];
}

// ---------------------------------------------------------------------------
// WMMA GEMM: D[M][Nn] = A[M][K](f32) @ Wpacked(bf16) + bias, opt relu.
// One wave per 16x16 tile; K stepped by 32 with v_wmma_f32_16x16x32_bf16.
// ---------------------------------------------------------------------------
__global__ void k_gemm_wmma(const float* __restrict__ A, const bf16_t* __restrict__ P,
                            const float* __restrict__ bias, float* __restrict__ D,
                            int M, int K, int Nn, int relu) {
  int mt = blockIdx.x, nt = blockIdx.y;
  int l = threadIdx.x;
  int ntiles = Nn >> 4;
  int row = mt * 16 + (l & 15);
  if (row >= M) row = M - 1;            // defensive clamp (all M here are /16)
  int kb = (l >> 4) << 4;               // 0 or 16 (A-matrix K base per lane half)
  v8f c = {0.f, 0.f, 0.f, 0.f, 0.f, 0.f, 0.f, 0.f};
  int ksteps = K >> 5;
  for (int kt = 0; kt < ksteps; ++kt) {
    const float* ap = A + (size_t)row * K + kt * 32 + kb;
    v16bf a, b;
#pragma unroll
    for (int j = 0; j < 8; ++j) { a[j] = (bf16_t)ap[j]; a[8 + j] = (bf16_t)ap[8 + j]; }
    const bf16_t* bp = P + ((size_t)(kt * ntiles + nt) * 32 + l) * 16;
#pragma unroll
    for (int j = 0; j < 16; ++j) b[j] = bp[j];
    c = __builtin_amdgcn_wmma_f32_16x16x32_bf16(false, a, false, b, (short)0, c,
                                                false, false);
  }
  int ncol = nt * 16 + (l & 15);
  float bi = bias ? bias[ncol] : 0.f;
  int mbase = mt * 16 + ((l >> 4) << 3);
#pragma unroll
  for (int r = 0; r < 8; ++r) {
    int mr = mbase + r;
    if (mr < M) {
      float v = c[r] + bi;
      if (relu) v = fmaxf(v, 0.f);
      D[(size_t)mr * Nn + ncol] = v;
    }
  }
}

// ---------------------------------------------------------------------------
// Self-loop mean edge feature: scatter e into fe tail + degree count
// ---------------------------------------------------------------------------
__global__ void k_eloop_scatter(const int* __restrict__ ei, const float* __restrict__ fe_e,
                                float* __restrict__ fe_loop, float* __restrict__ deg, int E) {
  int warp = (blockIdx.x * blockDim.x + threadIdx.x) >> 5;
  int lane = threadIdx.x & 31;
  if (warp >= E) return;
  int dst = ei[E + warp];
  const float* src = fe_e + (size_t)warp * HID;
  float* dp = fe_loop + (size_t)dst * HID;
#pragma unroll
  for (int r = 0; r < 4; ++r) atomicAdd(dp + r * 32 + lane, src[r * 32 + lane]);
  if (lane == 0) atomicAdd(deg + dst, 1.0f);
}

__global__ void k_eloop_div(float* __restrict__ loopfe, const float* __restrict__ deg, int Nn) {
  int idx = blockIdx.x * blockDim.x + threadIdx.x;
  if (idx >= Nn * HID) return;
  loopfe[idx] /= fmaxf(deg[idx >> 7], 1.f);
}

// ---------------------------------------------------------------------------
// a_s / a_d: per-head dot of xh row with attention vectors (warp reduce)
// ---------------------------------------------------------------------------
__global__ void k_att_node(const float* __restrict__ XH, const float* __restrict__ att_s,
                           const float* __restrict__ att_d, float* __restrict__ a_s,
                           float* __restrict__ a_d, int Nn) {
  int n = blockIdx.x;
  int t = threadIdx.x;                 // 128 threads = 4 warps = 4 heads
  int hh = t >> 5, c = t & 31;
  float v = XH[(size_t)n * HID + t];
  float ps = v * att_s[hh * CH + c];
  float pd = v * att_d[hh * CH + c];
  for (int off = 16; off > 0; off >>= 1) {
    ps += __shfl_down(ps, off, 32);
    pd += __shfl_down(pd, off, 32);
  }
  if (c == 0) { a_s[n * NHEAD + hh] = ps; a_d[n * NHEAD + hh] = pd; }
}

// Fold gat_edge_w with att_edge: wfold[k][h] = sum_c EW[k][h*32+c]*att_e[h][c]
__global__ void k_fold_attw(const float* __restrict__ EW, const float* __restrict__ att_e,
                            float* __restrict__ wfold) {
  int idx = blockIdx.x * blockDim.x + threadIdx.x;
  if (idx >= HID * NHEAD) return;
  int k = idx >> 2, hh = idx & 3;
  float s = 0.f;
  for (int c = 0; c < CH; ++c)
    s = fmaf(EW[(size_t)k * HID + hh * CH + c], att_e[hh * CH + c], s);
  wfold[k * NHEAD + hh] = s;
}

// a_e[e][h] = fe[e] . wfold[:,h]  (warp per edge, wfold staged in LDS)
__global__ void k_att_edge(const float* __restrict__ FE, const float* __restrict__ wfold,
                           float* __restrict__ a_e, int F) {
  __shared__ float ws[HID * NHEAD];
  for (int i = threadIdx.x; i < HID * NHEAD; i += blockDim.x) ws[i] = wfold[i];
  __syncthreads();
  int warp = (blockIdx.x * blockDim.x + threadIdx.x) >> 5;
  int lane = threadIdx.x & 31;
  if (warp >= F) return;
  const float* fp = FE + (size_t)warp * HID;
  float a0 = 0.f, a1 = 0.f, a2 = 0.f, a3 = 0.f;
#pragma unroll
  for (int r = 0; r < 4; ++r) {
    int k = r * 32 + lane;
    float v = fp[k];
    a0 = fmaf(v, ws[k * 4 + 0], a0);
    a1 = fmaf(v, ws[k * 4 + 1], a1);
    a2 = fmaf(v, ws[k * 4 + 2], a2);
    a3 = fmaf(v, ws[k * 4 + 3], a3);
  }
  for (int off = 16; off > 0; off >>= 1) {
    a0 += __shfl_down(a0, off, 32);
    a1 += __shfl_down(a1, off, 32);
    a2 += __shfl_down(a2, off, 32);
    a3 += __shfl_down(a3, off, 32);
  }
  if (lane == 0) {
    float* op = a_e + (size_t)warp * NHEAD;
    op[0] = a0; op[1] = a1; op[2] = a2; op[3] = a3;
  }
}

// alpha = leaky_relu(a_s[src]+a_d[dst]+a_e); segment max via ordered-uint atomicMax
__global__ void k_alpha(const int* __restrict__ ei, const float* __restrict__ a_s,
                        const float* __restrict__ a_d, const float* __restrict__ a_e,
                        float* __restrict__ alpha, unsigned* __restrict__ amax,
                        int E, int F) {
  int idx = blockIdx.x * blockDim.x + threadIdx.x;
  if (idx >= F * NHEAD) return;
  int e = idx >> 2, hh = idx & 3;
  int s = (e < E) ? ei[e] : (e - E);
  int d = (e < E) ? ei[E + e] : (e - E);
  float al = a_s[s * NHEAD + hh] + a_d[d * NHEAD + hh] + a_e[idx];
  al = (al >= 0.f) ? al : NEG_SLOPE * al;
  alpha[idx] = al;
  atomicMax(amax + d * NHEAD + hh, f2ord(al));
}

__global__ void k_expsum(const int* __restrict__ ei, const float* __restrict__ alpha,
                         const unsigned* __restrict__ amax, float* __restrict__ expa,
                         float* __restrict__ denom, int E, int F) {
  int idx = blockIdx.x * blockDim.x + threadIdx.x;
  if (idx >= F * NHEAD) return;
  int e = idx >> 2, hh = idx & 3;
  int d = (e < E) ? ei[E + e] : (e - E);
  float v = __expf(alpha[idx] - ord2f(amax[d * NHEAD + hh]));
  expa[idx] = v;
  atomicAdd(denom + d * NHEAD + hh, v);
}

// out[dst] += coef * xh[src]   (warp per edge, 4 channels/lane)
__global__ void k_agg(const int* __restrict__ ei, const float* __restrict__ XH,
                      const float* __restrict__ expa, const float* __restrict__ denom,
                      float* __restrict__ acc, int E, int F) {
  int warp = (blockIdx.x * blockDim.x + threadIdx.x) >> 5;
  int lane = threadIdx.x & 31;
  if (warp >= F) return;
  int e = warp;
  int s = (e < E) ? ei[e] : (e - E);
  int d = (e < E) ? ei[E + e] : (e - E);
  float coef[4];
#pragma unroll
  for (int r = 0; r < 4; ++r)
    coef[r] = expa[(size_t)e * NHEAD + r] / denom[(size_t)d * NHEAD + r];
  const float* xp = XH + (size_t)s * HID;
  float* ap = acc + (size_t)d * HID;
#pragma unroll
  for (int r = 0; r < 4; ++r) {
    int c = r * 32 + lane;
    atomicAdd(ap + c, xp[c] * coef[r]);
  }
}

// BN (eval) + relu + residual, in place on h
__global__ void k_bn_res(const float* __restrict__ acc, const float* __restrict__ gbias,
                         const float* __restrict__ gamma, const float* __restrict__ beta,
                         const float* __restrict__ mean, const float* __restrict__ var,
                         float* __restrict__ h, int Nn) {
  int idx = blockIdx.x * blockDim.x + threadIdx.x;
  if (idx >= Nn * HID) return;
  int c = idx & (HID - 1);
  float scale = gamma[c] * rsqrtf(var[c] + EPSV);
  float v = (acc[idx] + gbias[c] - mean[c]) * scale + beta[c];
  h[idx] = fmaxf(v, 0.f) + h[idx];
}

// multi-pool readout scatter: sum + ordered max + count
__global__ void k_pool(const float* __restrict__ H, const int* __restrict__ batch,
                       float* __restrict__ gsum, unsigned* __restrict__ gmaxu,
                       float* __restrict__ cnt, int Nn) {
  int warp = (blockIdx.x * blockDim.x + threadIdx.x) >> 5;
  int lane = threadIdx.x & 31;
  if (warp >= Nn) return;
  int b = batch[warp];
  const float* hp = H + (size_t)warp * HID;
#pragma unroll
  for (int r = 0; r < 4; ++r) {
    int c = r * 32 + lane;
    float v = hp[c];
    atomicAdd(gsum + (size_t)b * HID + c, v);
    atomicMax(gmaxu + (size_t)b * HID + c, f2ord(v));
  }
  if (lane == 0) atomicAdd(cnt + b, 1.0f);
}

// graph_in = [mean | max | sum]
__global__ void k_graph_build(const float* __restrict__ gsum, const unsigned* __restrict__ gmaxu,
                              const float* __restrict__ cnt, float* __restrict__ gin, int Bn) {
  int idx = blockIdx.x * blockDim.x + threadIdx.x;
  if (idx >= Bn * 3 * HID) return;
  int b = idx / (3 * HID), col = idx % (3 * HID);
  float cv = cnt[b];
  float out;
  if (col < HID) out = gsum[(size_t)b * HID + col] / fmaxf(cv, 1.f);
  else if (col < 2 * HID) {
    unsigned t = gmaxu[(size_t)b * HID + col - HID];
    out = (cv > 0.f) ? ord2f(t) : 0.f;
  } else out = gsum[(size_t)b * HID + col - 2 * HID];
  gin[idx] = out;
}

__global__ void k_concat2(const float* __restrict__ a, const float* __restrict__ b,
                          float* __restrict__ out, int Bn, int Na, int Nb) {
  int idx = blockIdx.x * blockDim.x + threadIdx.x;
  int tot = Na + Nb;
  if (idx >= Bn * tot) return;
  int r = idx / tot, c = idx % tot;
  out[idx] = (c < Na) ? a[(size_t)r * Na + c] : b[(size_t)r * Nb + (c - Na)];
}

// ---------------------------------------------------------------------------
extern "C" void kernel_launch(void* const* d_in, const int* in_sizes, int n_in,
                              void* d_out, int out_size, void* d_ws, size_t ws_size,
                              hipStream_t stream) {
  (void)n_in; (void)out_size; (void)ws_size;
  const float* x        = (const float*)d_in[0];
  const int*   ei       = (const int*)  d_in[1];
  const float* eattr    = (const float*)d_in[2];
  const int*   batch    = (const int*)  d_in[3];
  const float* gfeat    = (const float*)d_in[4];
  const float* node_w   = (const float*)d_in[5];
  const float* node_b   = (const float*)d_in[6];
  const float* edge_w   = (const float*)d_in[7];
  const float* edge_b   = (const float*)d_in[8];
  const float* gat_lin_w  = (const float*)d_in[9];
  const float* gat_edge_w = (const float*)d_in[10];
  const float* att_src  = (const float*)d_in[11];
  const float* att_dst  = (const float*)d_in[12];
  const float* att_edge = (const float*)d_in[13];
  const float* gat_bias = (const float*)d_in[14];
  const float* bn_gamma = (const float*)d_in[15];
  const float* bn_beta  = (const float*)d_in[16];
  const float* bn_mean  = (const float*)d_in[17];
  const float* bn_var   = (const float*)d_in[18];
  const float* gc_w  = (const float*)d_in[19];
  const float* gc_b  = (const float*)d_in[20];
  const float* gf1_w = (const float*)d_in[21];
  const float* gf1_b = (const float*)d_in[22];
  const float* gf2_w = (const float*)d_in[23];
  const float* gf2_b = (const float*)d_in[24];
  const float* p1_w  = (const float*)d_in[25];
  const float* p1_b  = (const float*)d_in[26];
  const float* p2_w  = (const float*)d_in[27];
  const float* p2_b  = (const float*)d_in[28];
  const float* p3_w  = (const float*)d_in[29];
  const float* p3_b  = (const float*)d_in[30];

  const int N  = in_sizes[0] / 7;
  const int E  = in_sizes[1] / 2;
  const int Bn = in_sizes[4] / GDIM;
  const int F  = E + N;

  // ---- carve workspace ----
  char* wsp = (char*)d_ws;
  size_t off = 0;
  auto carve = [&](size_t bytes) -> void* {
    void* p = wsp + off;
    off += (bytes + 255) & ~(size_t)255;
    return p;
  };
  float*    fe     = (float*)   carve((size_t)F * HID * 4);
  float*    h      = (float*)   carve((size_t)N * HID * 4);
  float*    xh     = (float*)   carve((size_t)N * HID * 4);
  float*    oacc   = (float*)   carve((size_t)N * HID * 4);
  float*    a_s    = (float*)   carve((size_t)N * NHEAD * 4);
  float*    a_d    = (float*)   carve((size_t)N * NHEAD * 4);
  float*    a_e    = (float*)   carve((size_t)F * NHEAD * 4);
  float*    alpha  = (float*)   carve((size_t)F * NHEAD * 4);
  float*    expa   = (float*)   carve((size_t)F * NHEAD * 4);
  unsigned* amaxu  = (unsigned*)carve((size_t)N * NHEAD * 4);
  float*    denom  = (float*)   carve((size_t)N * NHEAD * 4);
  float*    deg    = (float*)   carve((size_t)N * 4);
  float*    wfold  = (float*)   carve((size_t)HID * NHEAD * 4);
  bf16_t*   wpack  = (bf16_t*)  carve((size_t)3 * HID * HID * 2);
  float*    cnt    = (float*)   carve((size_t)Bn * 4);
  float*    gsum   = (float*)   carve((size_t)Bn * HID * 4);
  unsigned* gmaxu  = (unsigned*)carve((size_t)Bn * HID * 4);
  float*    gin    = (float*)   carve((size_t)Bn * 3 * HID * 4);
  float*    graphv = (float*)   carve((size_t)Bn * HID * 4);
  float*    glob1  = (float*)   carve((size_t)Bn * 64 * 4);
  float*    globv  = (float*)   carve((size_t)Bn * 64 * 4);
  float*    comb   = (float*)   carve((size_t)Bn * 192 * 4);
  float*    hid1   = (float*)   carve((size_t)Bn * HID * 4);
  float*    hid2   = (float*)   carve((size_t)Bn * 64 * 4);

  // ---- embeddings ----
  k_lin_small<<<N, 128, 7 * 4, stream>>>(x, node_w, node_b, h, 7, HID, 0);
  k_lin_small<<<E, 128, 3 * 4, stream>>>(eattr, edge_w, edge_b, fe, 3, HID, 0);

  // ---- self-loop mean edge features into fe tail ----
  hipMemsetAsync(fe + (size_t)E * HID, 0, (size_t)N * HID * 4, stream);
  hipMemsetAsync(deg, 0, (size_t)N * 4, stream);
  k_eloop_scatter<<<(E + 7) / 8, 256, 0, stream>>>(ei, fe, fe + (size_t)E * HID, deg, E);
  k_eloop_div<<<(N * HID + 255) / 256, 256, 0, stream>>>(fe + (size_t)E * HID, deg, N);

  // ---- GAT layers ----
  for (int L = 0; L < NLAYER; ++L) {
    const float* lw  = gat_lin_w  + (size_t)L * HID * HID;
    const float* ew  = gat_edge_w + (size_t)L * HID * HID;
    const float* asr = att_src  + (size_t)L * NHEAD * CH;
    const float* ads = att_dst  + (size_t)L * NHEAD * CH;
    const float* aed = att_edge + (size_t)L * NHEAD * CH;
    const float* gb  = gat_bias + (size_t)L * HID;

    k_pack_w<<<dim3(HID / 32, HID / 16), 32, 0, stream>>>(lw, wpack, HID, HID);
    k_gemm_wmma<<<dim3((N + 15) / 16, HID / 16), 32, 0, stream>>>(h, wpack, nullptr, xh,
                                                                  N, HID, HID, 0);
    k_att_node<<<N, 128, 0, stream>>>(xh, asr, ads, a_s, a_d, N);
    k_fold_attw<<<2, 256, 0, stream>>>(ew, aed, wfold);
    k_att_edge<<<(F + 7) / 8, 256, 0, stream>>>(fe, wfold, a_e, F);

    hipMemsetAsync(amaxu, 0, (size_t)N * NHEAD * 4, stream);
    k_alpha<<<(F * NHEAD + 255) / 256, 256, 0, stream>>>(ei, a_s, a_d, a_e, alpha, amaxu, E, F);
    hipMemsetAsync(denom, 0, (size_t)N * NHEAD * 4, stream);
    k_expsum<<<(F * NHEAD + 255) / 256, 256, 0, stream>>>(ei, alpha, amaxu, expa, denom, E, F);
    hipMemsetAsync(oacc, 0, (size_t)N * HID * 4, stream);
    k_agg<<<(F + 7) / 8, 256, 0, stream>>>(ei, xh, expa, denom, oacc, E, F);

    k_bn_res<<<(N * HID + 255) / 256, 256, 0, stream>>>(
        oacc, gb, bn_gamma + (size_t)L * HID, bn_beta + (size_t)L * HID,
        bn_mean + (size_t)L * HID, bn_var + (size_t)L * HID, h, N);
  }

  // ---- readout ----
  hipMemsetAsync(cnt, 0, (size_t)Bn * 4, stream);
  hipMemsetAsync(gsum, 0, (size_t)Bn * HID * 4, stream);
  hipMemsetAsync(gmaxu, 0, (size_t)Bn * HID * 4, stream);
  k_pool<<<(N + 7) / 8, 256, 0, stream>>>(h, batch, gsum, gmaxu, cnt, N);
  k_graph_build<<<(Bn * 3 * HID + 255) / 256, 256, 0, stream>>>(gsum, gmaxu, cnt, gin, Bn);

  k_pack_w<<<dim3(3 * HID / 32, HID / 16), 32, 0, stream>>>(gc_w, wpack, 3 * HID, HID);
  k_gemm_wmma<<<dim3((Bn + 15) / 16, HID / 16), 32, 0, stream>>>(gin, wpack, gc_b, graphv,
                                                                 Bn, 3 * HID, HID, 1);

  k_lin_small<<<Bn, 64, GDIM * 4, stream>>>(gfeat, gf1_w, gf1_b, glob1, GDIM, 64, 1);
  k_lin_small<<<Bn, 64, 64 * 4, stream>>>(glob1, gf2_w, gf2_b, globv, 64, 64, 0);

  k_concat2<<<(Bn * 192 + 255) / 256, 256, 0, stream>>>(graphv, globv, comb, Bn, HID, 64);

  k_pack_w<<<dim3(192 / 32, HID / 16), 32, 0, stream>>>(p1_w, wpack, 192, HID);
  k_gemm_wmma<<<dim3((Bn + 15) / 16, HID / 16), 32, 0, stream>>>(comb, wpack, p1_b, hid1,
                                                                 Bn, 192, HID, 1);
  k_pack_w<<<dim3(HID / 32, 64 / 16), 32, 0, stream>>>(p2_w, wpack, HID, 64);
  k_gemm_wmma<<<dim3((Bn + 15) / 16, 64 / 16), 32, 0, stream>>>(hid1, wpack, p2_b, hid2,
                                                                Bn, HID, 64, 1);
  k_lin_small<<<Bn, 64, 64 * 4, stream>>>(hid2, p3_w, p3_b, (float*)d_out, 64, TOUT, 0);
}
